// HybridCNN_QNN_90735479095377
// MI455X (gfx1250) — compile-verified, matrix operand order checked by reading, not run on hardware
//
#include <hip/hip_runtime.h>
#include <hip/hip_bf16.h>
#include <math.h>

typedef __attribute__((ext_vector_type(16))) _Float16 v16h;
typedef __attribute__((ext_vector_type(8)))  _Float16 v8h;
typedef __attribute__((ext_vector_type(4)))  _Float16 v4h;
typedef __attribute__((ext_vector_type(8)))  float    v8f;

#define BATCH 256
#define PI_F 3.14159265358979323846f
// 1/sqrt(1 + 1e-5) : eval-mode BN with running stats (0,1)
#define BN_INV 0.99999500003749968754f

// K index (within a 32-wide chunk) held by vector element j of a lane in
// half-block hb, per the CDNA5 16-bit WMMA A/B operand layout
// (CDNA5 ISA 7.12.2): lanes 0-15 -> K 0..7 / 16..23,
// lanes 16-31 -> K 8..15 / 24..31; 2 f16 per VGPR.
// Note: j in [0,8)  -> K = hb*8 + j        (contiguous)
//       j in [8,16) -> K = 16 + hb*8 + j-8 (contiguous)
__device__ __forceinline__ int wmma_k_of(int hb, int j) {
    int v = j >> 1, e = j & 1;
    return (v < 4) ? (hb * 8 + v * 2 + e) : (16 + hb * 8 + (v - 4) * 2 + e);
}

// ---------------------------------------------------------------------------
// Pack a [R x K] f32 matrix into WMMA operand layout:
// [tile(16 rows)][chunk(32 K)][lane(32)][16 f16], zero padded.
// conv_cin == 0 : src is row-major [R][K]                (plain GEMM operand)
// conv_cin >  0 : conv weights W[R][CIN][7], K-dim k-major: K = kk*CIN + ci
//                 (matches the position-major LDS im2col in the conv kernel)
// ---------------------------------------------------------------------------
__global__ __launch_bounds__(256)
void pack_wmma_kernel(const float* __restrict__ src, _Float16* __restrict__ dst,
                      int R, int K, int conv_cin) {
    int nch   = (K + 31) >> 5;
    int ntile = (R + 15) >> 4;
    int total = ntile * nch * 512;
    int idx = blockIdx.x * 256 + threadIdx.x;
    if (idx >= total) return;
    int j    = idx & 15;
    int lane = (idx >> 4) & 31;
    int rest = idx >> 9;
    int c    = rest % nch;
    int tile = rest / nch;
    int hb = lane >> 4;
    int r  = tile * 16 + (lane & 15);
    int k  = c * 32 + wmma_k_of(hb, j);
    _Float16 val = (_Float16)0.0f;
    if (r < R && k < K) {
        size_t sidx;
        if (conv_cin > 0) {
            int kk = k / conv_cin;          // CIN is a power of two -> shift
            int ci = k - kk * conv_cin;
            sidx = ((size_t)r * conv_cin + ci) * 7 + kk;
        } else {
            sidx = (size_t)r * K + k;
        }
        val = (_Float16)src[sidx];
    }
    dst[idx] = val;
}

// Fold conv bias + eval-mode BN into per-channel scale/shift:
// y = conv*sc + sh, sc = g/sqrt(1+eps), sh = bias*sc + beta
__global__ void bn_prep_kernel(const float* __restrict__ bias,
                               const float* __restrict__ gamma,
                               const float* __restrict__ beta,
                               float* __restrict__ sc, float* __restrict__ sh, int C) {
    int i = blockIdx.x * blockDim.x + threadIdx.x;
    if (i >= C) return;
    float s = gamma[i] * (float)BN_INV;
    sc[i] = s;
    sh[i] = bias[i] * s + beta[i];
}

// ---------------------------------------------------------------------------
// Fused Conv1d(K=7,pad=3) + bias + BN(eval) + ReLU via WMMA f16 -> f32.
// GEMM view: M = Cout (packed A), N = 16 output positions,
// K-dim = CIN*7 ordered k-major so that with the position-major LDS tile
// xs[p*CIN + ci] the B operand is LINEAR in K:  B[K][n] = xs[n*CIN + K].
// => per chunk each lane issues two contiguous 16B LDS loads, no index math.
// Block = 256 threads = 8 waves = (COUT/16) cout-tiles x (BP/16) pos-subtiles.
// ---------------------------------------------------------------------------
template<int CIN, int COUT, int BP, bool IN_F32>
__global__ __launch_bounds__(256)
void conv_bn_relu_wmma(const void* __restrict__ in_,
                       const _Float16* __restrict__ wpacked,
                       const float* __restrict__ sc,
                       const float* __restrict__ sh,
                       _Float16* __restrict__ out, int L) {
    constexpr int KD     = CIN * 7;
    constexpr int NCHK   = (KD + 31) / 32;
    constexpr int CTILES = COUT / 16;
    constexpr int TW     = BP + 6;          // halo = 6
    constexpr int XSZ    = CIN * TW + 32;   // pad so last-chunk reads stay in-bounds

    __shared__ _Float16 xs[XSZ];

    const int b   = blockIdx.y;
    const int tid = threadIdx.x;
    const int t0  = blockIdx.x * BP;

    const float*    inF = (const float*)in_;
    const _Float16* inH = (const _Float16*)in_;

    // Stage input tile into LDS, position-major, zero padded.
    // Read order is t-contiguous per channel (coalesced); write scatters.
    for (int i = tid; i < CIN * TW; i += 256) {
        int ci = i / TW, p = i - ci * TW;
        int t  = t0 - 3 + p;
        float v = 0.0f;
        if (t >= 0 && t < L) {
            size_t idx = ((size_t)b * CIN + ci) * L + t;
            v = IN_F32 ? inF[idx] : (float)inH[idx];
        }
        xs[p * CIN + ci] = (_Float16)v;
    }
    for (int i = CIN * TW + tid; i < XSZ; i += 256) xs[i] = (_Float16)0.0f;
    // Prefetch next position tile for this channel (global_prefetch_b8).
    if (tid < CIN && t0 + BP < L) {
        size_t nidx = ((size_t)b * CIN + tid) * L + (t0 + BP);
        if (IN_F32) __builtin_prefetch(inF + nidx, 0, 1);
        else        __builtin_prefetch(inH + nidx, 0, 1);
    }
    __syncthreads();

    const int wave = tid >> 5, lane = tid & 31;
    const int ct = wave % CTILES;          // cout tile
    const int ps = wave / CTILES;          // position sub-tile
    const int hb = lane >> 4, ln = lane & 15;
    const int pofs = ps * 16 + ln;         // column n = position within block

    v8f acc = {0.f, 0.f, 0.f, 0.f, 0.f, 0.f, 0.f, 0.f};
    const v16h* pa = (const v16h*)wpacked;
    const _Float16* bbase = xs + pofs * CIN;
    const int hb8 = hb * 8;

    #pragma unroll 4
    for (int c = 0; c < NCHK; ++c) {
        v16h av = pa[((size_t)ct * NCHK + c) * 32 + lane];  // packed weights
        v16h bv;
        const _Float16* bp = bbase + c * 32 + hb8;
        if constexpr (CIN >= 8) {           // 16B-aligned: two ds_load_b128
            v8h lo = *(const v8h*)(bp);
            v8h hi = *(const v8h*)(bp + 16);
            #pragma unroll
            for (int j = 0; j < 8; ++j) { bv[j] = lo[j]; bv[j + 8] = hi[j]; }
        } else {                            // conv1: 8B-aligned ds_load_b64 x4
            v4h a0 = *(const v4h*)(bp);
            v4h a1 = *(const v4h*)(bp + 4);
            v4h a2 = *(const v4h*)(bp + 16);
            v4h a3 = *(const v4h*)(bp + 20);
            #pragma unroll
            for (int j = 0; j < 4; ++j) {
                bv[j] = a0[j]; bv[j + 4] = a1[j];
                bv[j + 8] = a2[j]; bv[j + 12] = a3[j];
            }
        }
        acc = __builtin_amdgcn_wmma_f32_16x16x32_f16(
                false, av, false, bv, (short)0, acc, false, false);
    }

    // Epilogue: y = conv*sc + sh, ReLU, store f16.
    int t = t0 + pofs;
    if (t < L) {
        #pragma unroll
        for (int v = 0; v < 8; ++v) {
            int co  = ct * 16 + hb8 + v;            // D row M = v + 8*hb
            float y = acc[v] * sc[co] + sh[co];
            y = y > 0.f ? y : 0.f;
            out[((size_t)b * COUT + co) * L + t] = (_Float16)y;
        }
    }
}

// ---------------------------------------------------------------------------
// MaxPool1d(kernel=4, stride=4), f16 -> f16
// ---------------------------------------------------------------------------
__global__ __launch_bounds__(256)
void maxpool4_kernel(const _Float16* __restrict__ in, _Float16* __restrict__ out,
                     int Lin, int Lout, int total) {
    int i = blockIdx.x * 256 + threadIdx.x;
    if (i >= total) return;
    int t  = i % Lout;
    int bc = i / Lout;
    const _Float16* p = in + (size_t)bc * Lin + (size_t)t * 4;
    float m = (float)p[0];
    #pragma unroll
    for (int k = 1; k < 4; ++k) { float v = (float)p[k]; m = v > m ? v : m; }
    out[i] = (_Float16)m;
}

// ---------------------------------------------------------------------------
// Mean over last dim (AdaptiveAvgPool1d(1)), f16 -> f16
// ---------------------------------------------------------------------------
__global__ __launch_bounds__(256)
void mean_l_kernel(const _Float16* __restrict__ in, _Float16* __restrict__ out,
                   int Lin, int total) {
    int i = blockIdx.x * 256 + threadIdx.x;
    if (i >= total) return;
    const _Float16* p = in + (size_t)i * Lin;
    float s = 0.f;
    for (int t = 0; t < Lin; ++t) s += (float)p[t];
    out[i] = (_Float16)(s / (float)Lin);
}

// ---------------------------------------------------------------------------
// FC 128->128 + ReLU via WMMA. One block per 16-batch tile; 8 waves = 8
// output-feature tiles. K = 128 (4 chunks).
// ---------------------------------------------------------------------------
__global__ __launch_bounds__(256)
void fc_relu_wmma(const _Float16* __restrict__ hin,
                  const _Float16* __restrict__ wpacked,
                  const float* __restrict__ bias, _Float16* __restrict__ out) {
    __shared__ _Float16 hs[16 * 128];
    const int bt  = blockIdx.x;
    const int tid = threadIdx.x;
    for (int i = tid; i < 16 * 128; i += 256)
        hs[i] = hin[(size_t)(bt * 16 + (i >> 7)) * 128 + (i & 127)];
    __syncthreads();

    const int wave = tid >> 5, lane = tid & 31;
    const int hb = lane >> 4, ln = lane & 15;
    v8f acc = {0.f, 0.f, 0.f, 0.f, 0.f, 0.f, 0.f, 0.f};
    const v16h* pb = (const v16h*)wpacked;
    #pragma unroll
    for (int c = 0; c < 4; ++c) {
        v16h av;                      // A: batch rows from LDS
        #pragma unroll
        for (int j = 0; j < 16; ++j)
            av[j] = hs[ln * 128 + c * 32 + wmma_k_of(hb, j)];
        v16h bv = pb[((size_t)wave * 4 + c) * 32 + lane];   // B: packed fc_w
        acc = __builtin_amdgcn_wmma_f32_16x16x32_f16(
                false, av, false, bv, (short)0, acc, false, false);
    }
    #pragma unroll
    for (int v = 0; v < 8; ++v) {
        int mb = bt * 16 + hb * 8 + v;   // batch row
        int n  = wave * 16 + ln;         // output feature
        float y = acc[v] + bias[n];
        y = y > 0.f ? y : 0.f;
        out[(size_t)mb * 128 + n] = (_Float16)y;
    }
}

// ---------------------------------------------------------------------------
// Angle layer: raw = h @ ang_w.T + b (N=8 padded to 16), angles = pi*tanh(raw)
// One wave per 16-batch tile.
// ---------------------------------------------------------------------------
__global__ __launch_bounds__(32)
void ang_wmma(const _Float16* __restrict__ hin,
              const _Float16* __restrict__ wpacked,
              const float* __restrict__ bias, float* __restrict__ angles) {
    __shared__ _Float16 hs[16 * 128];
    const int bt  = blockIdx.x;
    const int tid = threadIdx.x;
    for (int i = tid; i < 16 * 128; i += 32)
        hs[i] = hin[(size_t)(bt * 16 + (i >> 7)) * 128 + (i & 127)];
    __syncthreads();

    const int hb = tid >> 4, ln = tid & 15;
    v8f acc = {0.f, 0.f, 0.f, 0.f, 0.f, 0.f, 0.f, 0.f};
    const v16h* pb = (const v16h*)wpacked;
    #pragma unroll
    for (int c = 0; c < 4; ++c) {
        v16h av;
        #pragma unroll
        for (int j = 0; j < 16; ++j)
            av[j] = hs[ln * 128 + c * 32 + wmma_k_of(hb, j)];
        v16h bv = pb[(size_t)c * 32 + tid];
        acc = __builtin_amdgcn_wmma_f32_16x16x32_f16(
                false, av, false, bv, (short)0, acc, false, false);
    }
    if (ln < 8) {
        #pragma unroll
        for (int v = 0; v < 8; ++v) {
            int mb = bt * 16 + hb * 8 + v;
            float raw = acc[v] + bias[ln];
            angles[(size_t)mb * 8 + ln] = PI_F * tanhf(raw);
        }
    }
}

// ---------------------------------------------------------------------------
// 8-qubit statevector sim: AngleEmbedding(RY) + 2 StronglyEntanglingLayers.
// One block (128 threads) per batch item; 256 complex amps in LDS.
// Wire w <-> bit (7-w) of the amplitude index (PennyLane MSB convention).
// ---------------------------------------------------------------------------
__global__ __launch_bounds__(128)
void qsim_kernel(const float* __restrict__ angles, const float* __restrict__ qw,
                 float* __restrict__ qz) {
    __shared__ float sre[256], sim_[256], spr[256];
    const int b = blockIdx.x, tid = threadIdx.x;

    for (int i = tid; i < 256; i += 128) { sre[i] = (i == 0) ? 1.f : 0.f; sim_[i] = 0.f; }
    __syncthreads();

    // AngleEmbedding: RY(angle) on each wire (real 2x2)
    for (int w = 0; w < 8; ++w) {
        float a = angles[(size_t)b * 8 + w] * 0.5f;
        float c = cosf(a), s = sinf(a);
        int bp = 7 - w, mk = 1 << bp;
        int i0 = ((tid >> bp) << (bp + 1)) | (tid & (mk - 1));
        int i1 = i0 | mk;
        float r0 = sre[i0], q0 = sim_[i0], r1 = sre[i1], q1 = sim_[i1];
        sre[i0] = c * r0 - s * r1;  sim_[i0] = c * q0 - s * q1;
        sre[i1] = s * r0 + c * r1;  sim_[i1] = s * q0 + c * q1;
        __syncthreads();
    }

    // StronglyEntanglingLayers
    for (int l = 0; l < 2; ++l) {
        for (int w = 0; w < 8; ++w) {
            float phi = qw[(l * 8 + w) * 3 + 0];
            float th  = qw[(l * 8 + w) * 3 + 1];
            float om  = qw[(l * 8 + w) * 3 + 2];
            float cm = cosf(th * 0.5f), sm = sinf(th * 0.5f);
            float ap = -0.5f * (phi + om);   // m00 = e^{i*ap} cm, m11 = conj
            float am =  0.5f * (phi - om);   // m01 = -e^{i*am} sm, m10 = e^{-i*am} sm
            float m00r =  cosf(ap) * cm, m00i =  sinf(ap) * cm;
            float m01r = -cosf(am) * sm, m01i = -sinf(am) * sm;
            float m10r =  cosf(am) * sm, m10i = -sinf(am) * sm;
            float m11r =  cosf(ap) * cm, m11i = -sinf(ap) * cm;
            int bp = 7 - w, mk = 1 << bp;
            int i0 = ((tid >> bp) << (bp + 1)) | (tid & (mk - 1));
            int i1 = i0 | mk;
            float r0 = sre[i0], q0 = sim_[i0], r1 = sre[i1], q1 = sim_[i1];
            sre[i0]  = m00r * r0 - m00i * q0 + m01r * r1 - m01i * q1;
            sim_[i0] = m00r * q0 + m00i * r0 + m01r * q1 + m01i * r1;
            sre[i1]  = m10r * r0 - m10i * q0 + m11r * r1 - m11i * q1;
            sim_[i1] = m10r * q0 + m10i * r0 + m11r * q1 + m11i * r1;
            __syncthreads();
        }
        int r = (l % 7) + 1;
        for (int i = 0; i < 8; ++i) {
            int cq = i, tq = (i + r) & 7;
            int bc = 7 - cq, btg = 7 - tq;
            if (tid < 64) {  // 64 disjoint swap pairs in control=1 subspace
                int lo = bc < btg ? bc : btg, hi = bc < btg ? btg : bc;
                int p1 = ((tid >> lo) << (lo + 1)) | (tid & ((1 << lo) - 1));
                int p2 = ((p1 >> hi) << (hi + 1)) | (p1 & ((1 << hi) - 1));
                int ia = p2 | (1 << bc);     // control=1, target=0
                int ib = ia | (1 << btg);    // target=1
                float tr = sre[ia], tq2 = sim_[ia];
                sre[ia] = sre[ib];  sim_[ia] = sim_[ib];
                sre[ib] = tr;       sim_[ib] = tq2;
            }
            __syncthreads();
        }
    }

    for (int i = tid; i < 256; i += 128) spr[i] = sre[i] * sre[i] + sim_[i] * sim_[i];
    __syncthreads();
    if (tid < 8) {
        int bp = 7 - tid;
        float z = 0.f;
        for (int i = 0; i < 256; ++i)
            z += spr[i] * (((i >> bp) & 1) ? -1.f : 1.f);
        qz[(size_t)b * 8 + tid] = z;
    }
}

// ---------------------------------------------------------------------------
// Head MLP: relu(q @ h1_w.T + h1_b) @ h2_w.T + h2_b -> (256, 5) f32
// ---------------------------------------------------------------------------
__global__ __launch_bounds__(64)
void head_kernel(const float* __restrict__ qz,
                 const float* __restrict__ h1w, const float* __restrict__ h1b,
                 const float* __restrict__ h2w, const float* __restrict__ h2b,
                 float* __restrict__ out) {
    __shared__ float h1[64];
    int b = blockIdx.x, j = threadIdx.x;
    float s = h1b[j];
    #pragma unroll
    for (int k = 0; k < 8; ++k) s += qz[(size_t)b * 8 + k] * h1w[j * 8 + k];
    h1[j] = s > 0.f ? s : 0.f;
    __syncthreads();
    if (j < 5) {
        float o = h2b[j];
        for (int k = 0; k < 64; ++k) o += h1[k] * h2w[j * 64 + k];
        out[(size_t)b * 5 + j] = o;
    }
}

// ---------------------------------------------------------------------------
extern "C" void kernel_launch(void* const* d_in, const int* in_sizes, int n_in,
                              void* d_out, int out_size, void* d_ws, size_t ws_size,
                              hipStream_t stream) {
    (void)in_sizes; (void)n_in; (void)out_size; (void)ws_size;
    const float* x    = (const float*)d_in[0];
    const float* c1w  = (const float*)d_in[1];  const float* c1b = (const float*)d_in[2];
    const float* g1   = (const float*)d_in[3];  const float* be1 = (const float*)d_in[4];
    const float* c2w  = (const float*)d_in[5];  const float* c2b = (const float*)d_in[6];
    const float* g2   = (const float*)d_in[7];  const float* be2 = (const float*)d_in[8];
    const float* c3w  = (const float*)d_in[9];  const float* c3b = (const float*)d_in[10];
    const float* g3   = (const float*)d_in[11]; const float* be3 = (const float*)d_in[12];
    const float* c4w  = (const float*)d_in[13]; const float* c4b = (const float*)d_in[14];
    const float* g4   = (const float*)d_in[15]; const float* be4 = (const float*)d_in[16];
    const float* fcw  = (const float*)d_in[17]; const float* fcb = (const float*)d_in[18];
    const float* angw = (const float*)d_in[19]; const float* angb= (const float*)d_in[20];
    const float* qw   = (const float*)d_in[21];
    const float* h1w  = (const float*)d_in[22]; const float* h1b = (const float*)d_in[23];
    const float* h2w  = (const float*)d_in[24]; const float* h2b = (const float*)d_in[25];

    char* ws = (char*)d_ws;
    size_t off = 0;
    auto alloc = [&](size_t bytes) -> void* {
        void* p = ws + off;
        off = (off + bytes + 255) & ~(size_t)255;
        return p;
    };

    // Packed weights (f16, WMMA operand layout)
    _Float16* pk1   = (_Float16*)alloc((size_t)2  * 1  * 512 * 2);  // 32 x 28
    _Float16* pk2   = (_Float16*)alloc((size_t)4  * 7  * 512 * 2);  // 64 x 224
    _Float16* pk3   = (_Float16*)alloc((size_t)8  * 14 * 512 * 2);  // 128 x 448
    _Float16* pk4   = (_Float16*)alloc((size_t)8  * 28 * 512 * 2);  // 128 x 896
    _Float16* pkfc  = (_Float16*)alloc((size_t)8  * 4  * 512 * 2);  // 128 x 128
    _Float16* pkang = (_Float16*)alloc((size_t)1  * 4  * 512 * 2);  // 8 x 128 (pad)
    // Folded BN scale/shift per layer
    float* sc1 = (float*)alloc(32  * 4); float* sh1 = (float*)alloc(32  * 4);
    float* sc2 = (float*)alloc(64  * 4); float* sh2 = (float*)alloc(64  * 4);
    float* sc3 = (float*)alloc(128 * 4); float* sh3 = (float*)alloc(128 * 4);
    float* sc4 = (float*)alloc(128 * 4); float* sh4 = (float*)alloc(128 * 4);
    // Ping-pong activation buffers (f16)
    _Float16* buf1  = (_Float16*)alloc((size_t)BATCH * 32  * 3000 * 2); // 49.2 MB
    _Float16* buf2  = (_Float16*)alloc((size_t)BATCH * 64  * 3000 * 2); // 98.3 MB
    _Float16* hmean = (_Float16*)alloc((size_t)BATCH * 128 * 2);
    _Float16* hfc   = (_Float16*)alloc((size_t)BATCH * 128 * 2);
    float*    angsb = (float*)   alloc((size_t)BATCH * 8 * 4);
    float*    qzb   = (float*)   alloc((size_t)BATCH * 8 * 4);

    // Pack all matmul weights into WMMA layout (conv packs use k-major K-dim).
    pack_wmma_kernel<<<(2*1*512 + 255) / 256, 256, 0, stream>>>(c1w,  pk1,   32,  28,   4);
    pack_wmma_kernel<<<(4*7*512 + 255) / 256, 256, 0, stream>>>(c2w,  pk2,   64, 224,  32);
    pack_wmma_kernel<<<(8*14*512 + 255) / 256, 256, 0, stream>>>(c3w, pk3,  128, 448,  64);
    pack_wmma_kernel<<<(8*28*512 + 255) / 256, 256, 0, stream>>>(c4w, pk4,  128, 896, 128);
    pack_wmma_kernel<<<(8*4*512 + 255) / 256, 256, 0, stream>>>(fcw,  pkfc, 128, 128,   0);
    pack_wmma_kernel<<<(1*4*512 + 255) / 256, 256, 0, stream>>>(angw, pkang,  8, 128,   0);
    bn_prep_kernel<<<1, 128, 0, stream>>>(c1b, g1, be1, sc1, sh1, 32);
    bn_prep_kernel<<<1, 128, 0, stream>>>(c2b, g2, be2, sc2, sh2, 64);
    bn_prep_kernel<<<1, 128, 0, stream>>>(c3b, g3, be3, sc3, sh3, 128);
    bn_prep_kernel<<<1, 128, 0, stream>>>(c4b, g4, be4, sc4, sh4, 128);

    // conv1: (256,4,3000) f32 -> (256,32,3000) f16   [BP=64 -> 47 blocks]
    conv_bn_relu_wmma<4, 32, 64, true>
        <<<dim3(47, BATCH), 256, 0, stream>>>(x, pk1, sc1, sh1, buf1, 3000);
    // conv2: -> (256,64,3000)                        [BP=32 -> 94 blocks]
    conv_bn_relu_wmma<32, 64, 32, false>
        <<<dim3(94, BATCH), 256, 0, stream>>>(buf1, pk2, sc2, sh2, buf2, 3000);
    // pool -> (256,64,750) into buf1
    {
        int total = BATCH * 64 * 750;
        maxpool4_kernel<<<(total + 255) / 256, 256, 0, stream>>>(buf2, buf1, 3000, 750, total);
    }
    // conv3: -> (256,128,750) into buf2              [BP=16 -> 47 blocks]
    conv_bn_relu_wmma<64, 128, 16, false>
        <<<dim3(47, BATCH), 256, 0, stream>>>(buf1, pk3, sc3, sh3, buf2, 750);
    // pool -> (256,128,187) into buf1
    {
        int total = BATCH * 128 * 187;
        maxpool4_kernel<<<(total + 255) / 256, 256, 0, stream>>>(buf2, buf1, 750, 187, total);
    }
    // conv4: -> (256,128,187) into buf2              [BP=16 -> 12 blocks]
    conv_bn_relu_wmma<128, 128, 16, false>
        <<<dim3(12, BATCH), 256, 0, stream>>>(buf1, pk4, sc4, sh4, buf2, 187);
    // global average pool -> (256,128)
    {
        int total = BATCH * 128;
        mean_l_kernel<<<(total + 255) / 256, 256, 0, stream>>>(buf2, hmean, 187, total);
    }
    // fc + relu -> (256,128)
    fc_relu_wmma<<<16, 256, 0, stream>>>(hmean, pkfc, fcb, hfc);
    // angles = pi * tanh(h @ ang_w.T + b) -> (256,8)
    ang_wmma<<<16, 32, 0, stream>>>(hfc, pkang, angb, angsb);
    // 8-qubit statevector simulation -> (256,8) expvals
    qsim_kernel<<<BATCH, 128, 0, stream>>>(angsb, qw, qzb);
    // head MLP -> (256,5) f32
    head_kernel<<<BATCH, 64, 0, stream>>>(qzb, h1w, h1b, h2w, h2b, (float*)d_out);
}
